// MultiNodeOutputLayer_35854386987248
// MI455X (gfx1250) — matrix operand order repaired and checked by reading, not compiled, not run on hardware
//
#include <hip/hip_runtime.h>

// Problem constants (reference: B=2048, S=128, D=512, O=128)
#define B_ 2048
#define S_ 128
#define D_ 512
#define O_ 128
#define K_ (S_ * D_)          // 65536 contraction length

#define SPLITK 16
#define KSLICE (K_ / SPLITK)  // 4096
#define KT 32                 // K per iteration (bf16 WMMA K)
#define ITERS (KSLICE / KT)   // 128
#define BM 64                 // block tile M
#define BN 64                 // block tile N
#define XSB 40                // LDS row stride in bf16 elems (80B: 16B-aligned rows,
                              // r*20 mod 64 distinct for r=0..15 -> conflict-free reads)

typedef __attribute__((ext_vector_type(16))) __bf16 v16bf;
typedef __attribute__((ext_vector_type(8)))  __bf16 v8bf;
typedef __attribute__((ext_vector_type(8)))  float  v8f;

// out[b,o] = sum_s bias[s,o]  (also fully initializes poisoned d_out)
__global__ __launch_bounds__(256)
void bias_init_kernel(const float* __restrict__ bias, float* __restrict__ out) {
    int idx = blockIdx.x * 256 + threadIdx.x;        // 0 .. B_*O_-1
    int o = idx & (O_ - 1);
    float acc = 0.0f;
#pragma unroll 8
    for (int s = 0; s < S_; ++s) acc += bias[s * O_ + o];
    out[idx] = acc;
}

__device__ __forceinline__ v8bf cvt8(const float4& a, const float4& b) {
    v8bf r;
    r[0] = (__bf16)a.x; r[1] = (__bf16)a.y; r[2] = (__bf16)a.z; r[3] = (__bf16)a.w;
    r[4] = (__bf16)b.x; r[5] = (__bf16)b.y; r[6] = (__bf16)b.z; r[7] = (__bf16)b.w;
    return r;
}

__global__ __launch_bounds__(256)
void gemm_kernel(const float* __restrict__ x, const float* __restrict__ w,
                 float* __restrict__ out) {
    __shared__ __bf16 Xlds[BM * XSB];
    __shared__ __bf16 Wlds[BN * XSB];

    const int bx = blockIdx.x;
    const int bn = bx & 1;            // N fastest -> X-tile L2 reuse across bn
    const int bm = (bx >> 1) & 31;
    const int ks = bx >> 6;           // split-K slice
    const int m0 = bm * BM;
    const int o0 = bn * BN;
    const int kbase = ks * KSLICE;

    const int t    = threadIdx.x;
    const int lane = t & 31;
    const int wv   = t >> 5;
    const int wm   = wv & 3;          // wave M sub-tile (16 rows)
    const int wn   = wv >> 2;         // wave N sub-tile (32 cols)
    const int nn   = lane & 15;
    const int hi   = lane >> 4;

    // global->LDS staging: 4 threads per row, 8 floats each
    const int xr = t >> 2;            // 0..63
    const int xq = (t & 3) * 8;       // 0,8,16,24 (elements)

    v8f c0 = {};
    v8f c1 = {};

    float4 xa0, xa1, wa0, wa1;

    // prologue: load K-step 0 into registers
    {
        const int k0 = kbase;
        const int s  = k0 >> 9;       // KT=32 divides D_=512: a k-step stays within one s
        const int d0 = k0 & (D_ - 1);
        const float* xp = x + (size_t)(m0 + xr) * K_ + k0 + xq;
        xa0 = *(const float4*)xp;
        xa1 = *(const float4*)(xp + 4);
        const float* wp = w + (size_t)s * (O_ * D_) + (size_t)(o0 + xr) * D_ + d0 + xq;
        wa0 = *(const float4*)wp;
        wa1 = *(const float4*)(wp + 4);
    }

#pragma unroll 1
    for (int it = 0; it < ITERS; ++it) {
        __syncthreads();
        // convert once at the producer: fp32 regs -> bf16 LDS (1 ds_store_b128 each)
        *(v8bf*)&Xlds[xr * XSB + xq] = cvt8(xa0, xa1);
        *(v8bf*)&Wlds[xr * XSB + xq] = cvt8(wa0, wa1);
        __syncthreads();

        if (it + 1 < ITERS) {
            const int k0 = kbase + (it + 1) * KT;
            const int s  = k0 >> 9;
            const int d0 = k0 & (D_ - 1);
            const float* xp = x + (size_t)(m0 + xr) * K_ + k0 + xq;
            xa0 = *(const float4*)xp;
            xa1 = *(const float4*)(xp + 4);
            const float* wp = w + (size_t)s * (O_ * D_) + (size_t)(o0 + xr) * D_ + d0 + xq;
            wa0 = *(const float4*)wp;
            wa1 = *(const float4*)(wp + 4);
            if (it + 2 < ITERS) {
                __builtin_prefetch(xp + KT, 0, 3);   // global_prefetch_b8
                __builtin_prefetch(wp + KT, 0, 3);
            }
        }

        // ---- A fragment (16-bit A 16x32): lane M=nn, elem e -> K = e + hi*16 ----
        const __bf16* Xa = &Xlds[(wm * 16 + nn) * XSB + hi * 16];
        const v8bf a0 = *(const v8bf*)(Xa);
        const v8bf a1 = *(const v8bf*)(Xa + 8);
        v16bf a;
#pragma unroll
        for (int e = 0; e < 8; ++e) { a[e] = a0[e]; a[e + 8] = a1[e]; }

        // ---- B fragments (16-bit B 32x16): lane N=nn, elem e -> K = e + hi*16 ----
        const __bf16* Wb0 = &Wlds[(wn * 32 + nn) * XSB + hi * 16];
        const __bf16* Wb1 = Wb0 + 16 * XSB;
        const v8bf b00 = *(const v8bf*)(Wb0);
        const v8bf b01 = *(const v8bf*)(Wb0 + 8);
        const v8bf b10 = *(const v8bf*)(Wb1);
        const v8bf b11 = *(const v8bf*)(Wb1 + 8);
        v16bf b0, b1;
#pragma unroll
        for (int e = 0; e < 8; ++e) {
            b0[e] = b00[e]; b0[e + 8] = b01[e];
            b1[e] = b10[e]; b1[e + 8] = b11[e];
        }

        c0 = __builtin_amdgcn_wmma_f32_16x16x32_bf16(false, a, false, b0,
                                                     (short)0, c0, false, false);
        c1 = __builtin_amdgcn_wmma_f32_16x16x32_bf16(false, a, false, b1,
                                                     (short)0, c1, false, false);
    }

    // ---- epilogue: C/D layout VGPR r -> M = r + hi*8, N = nn ----
    const int row  = m0 + wm * 16 + hi * 8;
    const int colb = o0 + wn * 32 + nn;
#pragma unroll
    for (int r = 0; r < 8; ++r) {
        atomicAdd(&out[(size_t)(row + r) * O_ + colb],      c0[r]);
        atomicAdd(&out[(size_t)(row + r) * O_ + colb + 16], c1[r]);
    }
}

extern "C" void kernel_launch(void* const* d_in, const int* in_sizes, int n_in,
                              void* d_out, int out_size, void* d_ws, size_t ws_size,
                              hipStream_t stream) {
    const float* x    = (const float*)d_in[0];  // [B, S, D]
    const float* w    = (const float*)d_in[1];  // [S, O, D]
    const float* bias = (const float*)d_in[2];  // [S, O]
    float* out        = (float*)d_out;          // [B, O]

    bias_init_kernel<<<(B_ * O_) / 256, 256, 0, stream>>>(bias, out);
    gemm_kernel<<<2 * 32 * SPLITK, 256, 0, stream>>>(x, w, out);
}
